// MinibatchDiscriminator64_68599217652003
// MI455X (gfx1250) — compile-verified, matrix operand order checked by reading, not run on hardware
//
#include <hip/hip_runtime.h>
#include <hip/hip_bf16.h>
#include <stdint.h>

// CDNA5 / gfx1250 wave32 WMMA types
typedef __attribute__((ext_vector_type(16))) __bf16 v16bf;
typedef __attribute__((ext_vector_type(8)))  __bf16 v8bf;
typedef __attribute__((ext_vector_type(8)))  float  v8f;

__device__ __forceinline__ v16bf ldfrag(const __bf16* p) {
  return *(const v16bf*)p;   // 32B contiguous -> 2x global_load_b128
}
__device__ __forceinline__ v16bf ldsfrag(const __bf16* p) {
  v8bf lo = *(const v8bf*)p;        // 16B LDS read
  v8bf hi = *(const v8bf*)(p + 8);  // 16B LDS read
  return __builtin_shufflevector(lo, hi, 0,1,2,3,4,5,6,7,8,9,10,11,12,13,14,15);
}
__device__ __forceinline__ v16bf zfrag() {
  __bf16 z = (__bf16)0.0f;
  v16bf v = { z,z,z,z,z,z,z,z,z,z,z,z,z,z,z,z };
  return v;
}
__device__ __forceinline__ v8f zacc() {
  v8f v = { 0.f,0.f,0.f,0.f,0.f,0.f,0.f,0.f };
  return v;
}

// gfx1250 async copy global->LDS (16B per lane), tracked by ASYNCcnt.
__device__ __forceinline__ void async_b128(uint32_t ldsaddr, const __bf16* gsrc) {
  asm volatile("global_load_async_to_lds_b128 %0, %1, off"
               :: "v"(ldsaddr), "v"(gsrc) : "memory");
}
__device__ __forceinline__ void wait_async0() {
  asm volatile("s_wait_asynccnt 0x0" ::: "memory");
}

#define BSTRIDE 40   // LDS row pitch in __bf16 (80B = 64B data + 16B pad -> bank-conflict-free)

#define WMMA_BF16(ACC, A_, B_) \
  ACC = __builtin_amdgcn_wmma_f32_16x16x32_bf16(false, A_, false, B_, (short)0, ACC, false, false)

// ---------------------------------------------------------------------------
// Layer-1 im2col: x (NCHW f32, 256x3x64x64) -> A1 (262144 x 96 bf16), K padded 75->96
__global__ void k_im2col1(const float* __restrict__ x, __bf16* __restrict__ A1) {
  size_t idx = (size_t)blockIdx.x * 256 + threadIdx.x;
  const size_t total = (size_t)262144 * 96;
  if (idx >= total) return;
  int k = (int)(idx % 96);
  size_t m = idx / 96;
  float v = 0.f;
  if (k < 75) {
    int ci = k % 3; int t = k / 3; int kw = t % 5; int kh = t / 5;
    int ow = (int)(m & 31); size_t t2 = m >> 5;
    int oh = (int)(t2 & 31); int n = (int)(t2 >> 5);
    int ih = oh*2 - 2 + kh, iw = ow*2 - 2 + kw;
    if ((unsigned)ih < 64u && (unsigned)iw < 64u)
      v = x[(((size_t)n*3 + ci)*64 + ih)*64 + iw];
  }
  A1[idx] = (__bf16)v;
}

// Weight transform: w (O,I,R,R) f32 OIHW -> Wt (O x Kpad) bf16, k=(kh*R+kw)*I+ci
__global__ void k_wt(const float* __restrict__ w, __bf16* __restrict__ Wt,
                     int I, int R, int Kpad, size_t total) {
  size_t idx = (size_t)blockIdx.x * 256 + threadIdx.x;
  if (idx >= total) return;
  int k = (int)(idx % Kpad);
  int o = (int)(idx / Kpad);
  float v = 0.f;
  int Kreal = R*R*I;
  if (k < Kreal) {
    int ci = k % I; int t = k / I; int kw = t % R; int kh = t / R;
    v = w[(((size_t)o*I + ci)*R + kh)*R + kw];
  }
  Wt[idx] = (__bf16)v;
}

// T (4096,64,50) f32 -> Tt[(o*50+k)][i] bf16 (3200 x 4096)
__global__ void k_transT(const float* __restrict__ T, __bf16* __restrict__ Tt) {
  size_t idx = (size_t)blockIdx.x * 256 + threadIdx.x;
  const size_t total = (size_t)3200 * 4096;
  if (idx >= total) return;
  int i  = (int)(idx & 4095);
  int ok = (int)(idx >> 12);
  Tt[idx] = (__bf16)T[(size_t)i*3200 + ok];
}

__global__ void k_cvt(const float* __restrict__ s, __bf16* __restrict__ d, size_t total) {
  size_t idx = (size_t)blockIdx.x * 256 + threadIdx.x;
  if (idx < total) d[idx] = (__bf16)s[idx];
}

// ---------------------------------------------------------------------------
// bf16 WMMA GEMM, async-LDS-staged B, double buffered, last-iteration peeled.
// C[M x N] f32 = A[M x K] * Bt[N x K]^T (+bias[n]).
// M % 256 == 0 (grid.x = M/256), K % 32 == 0, lda/ldb % 32 == 0.
// Block = 256 threads = 8 waves; block tile 256(M) x 64(N); each wave computes
// a 32x64 tile (8 WMMA accumulators), sharing the 4KB B tile staged in LDS.
__global__ void k_gemm(const __bf16* __restrict__ A, const __bf16* __restrict__ Bt,
                       float* __restrict__ C, const float* __restrict__ bias,
                       int N, int K, int lda, int ldb) {
  __shared__ __bf16 bs[2][64 * BSTRIDE];   // 2 x 5120 B

  int tid  = threadIdx.x;
  int lane = tid & 31;
  int wave = tid >> 5;
  int mt   = blockIdx.x * 8 + wave;            // 32-row tile index
  int m0   = mt*32 + (lane & 15);
  int koff = (lane >> 4) * 16;
  const __bf16* arow0 = A + (size_t)m0 * lda + koff;
  const __bf16* arow1 = arow0 + (size_t)16 * lda;

  // staging map: thread t -> B row n = t>>2, 16B chunk c = t&3 (row = 32 bf16 = 64B)
  int bn = tid >> 2, bc = tid & 3;
  int nglob = blockIdx.y*64 + bn; if (nglob > N-1) nglob = N-1;  // ragged N (fc)
  const __bf16* bsrc = Bt + (size_t)nglob * ldb + bc*8;
  uint32_t blds[2] = {
    (uint32_t)(uintptr_t)&bs[0][bn*BSTRIDE + bc*8],
    (uint32_t)(uintptr_t)&bs[1][bn*BSTRIDE + bc*8]
  };
  // per-lane fragment base in LDS: column (lane&15), k-sub (lane>>4)*16
  int foff = (lane & 15)*BSTRIDE + (lane >> 4)*16;

  v8f acc[8] = { zacc(), zacc(), zacc(), zacc(), zacc(), zacc(), zacc(), zacc() };

  int nt = K / 32;
  async_b128(blds[0], bsrc);                 // stage tile 0
  v16bf a0 = ldfrag(arow0);
  v16bf a1 = ldfrag(arow1);
  wait_async0();
  __syncthreads();

  // branch-free steady state: iterations 0 .. nt-2
  for (int i = 0; i < nt - 1; ++i) {
    async_b128(blds[(i+1)&1], bsrc + (size_t)(i+1)*32);   // DMA next tile

    const __bf16* bb = &bs[i & 1][foff];
    v16bf b0 = ldsfrag(bb);
    v16bf b1 = ldsfrag(bb + 16*BSTRIDE);
    v16bf b2 = ldsfrag(bb + 32*BSTRIDE);
    v16bf b3 = ldsfrag(bb + 48*BSTRIDE);

    v16bf a0n = ldfrag(arow0 + (size_t)(i+1)*32);         // pipeline A
    v16bf a1n = ldfrag(arow1 + (size_t)(i+1)*32);

    WMMA_BF16(acc[0], a0, b0); WMMA_BF16(acc[1], a0, b1);
    WMMA_BF16(acc[2], a0, b2); WMMA_BF16(acc[3], a0, b3);
    WMMA_BF16(acc[4], a1, b0); WMMA_BF16(acc[5], a1, b1);
    WMMA_BF16(acc[6], a1, b2); WMMA_BF16(acc[7], a1, b3);
    a0 = a0n; a1 = a1n;

    wait_async0();
    __syncthreads();                                       // next tile resident
  }

  // peeled last iteration (no async, no barrier, no A prefetch)
  {
    const __bf16* bb = &bs[(nt - 1) & 1][foff];
    v16bf b0 = ldsfrag(bb);
    v16bf b1 = ldsfrag(bb + 16*BSTRIDE);
    v16bf b2 = ldsfrag(bb + 32*BSTRIDE);
    v16bf b3 = ldsfrag(bb + 48*BSTRIDE);
    WMMA_BF16(acc[0], a0, b0); WMMA_BF16(acc[1], a0, b1);
    WMMA_BF16(acc[2], a0, b2); WMMA_BF16(acc[3], a0, b3);
    WMMA_BF16(acc[4], a1, b0); WMMA_BF16(acc[5], a1, b1);
    WMMA_BF16(acc[6], a1, b2); WMMA_BF16(acc[7], a1, b3);
  }

  // C/D layout: lane -> col = lane&15, rows = (lane>>4)*8 + vgpr
  int rbase0 = mt*32 + (lane >> 4) * 8;
  int col0   = blockIdx.y * 64 + (lane & 15);
  #pragma unroll
  for (int s = 0; s < 4; ++s) {
    int col = col0 + s*16;
    if (col < N) {
      float bv = bias ? bias[col] : 0.f;
      #pragma unroll
      for (int r = 0; r < 8; ++r) {
        C[(size_t)(rbase0 + r)      * N + col] = acc[s][r]     + bv;
        C[(size_t)(rbase0 + 16 + r) * N + col] = acc[4 + s][r] + bv;
      }
    }
  }
}

// ---------------------------------------------------------------------------
// Implicit-GEMM conv (layers 2-4): act NHWC bf16 -> out NHWC f32.
// Weights (B side) async-staged in LDS exactly like k_gemm; A gathered
// implicitly (Cin % 16 == 0 keeps each 16-wide K chunk inside one (kh,kw)).
// Cout % 64 == 0, (Nb*OH*OW) % 256 == 0.
__global__ void k_conv(const __bf16* __restrict__ act, const __bf16* __restrict__ Wt,
                       float* __restrict__ out,
                       int IH, int IW, int Cin, int OH, int OW, int Cout,
                       int R, int S, int P) {
  __shared__ __bf16 bs[2][64 * BSTRIDE];

  int tid  = threadIdx.x;
  int lane = tid & 31;
  int wave = tid >> 5;
  int mt   = blockIdx.x * 8 + wave;
  int m0   = mt*32 + (lane & 15);
  int m1   = m0 + 16;
  int koff = (lane >> 4) * 16;
  int K  = R * R * Cin;
  int RC = R * Cin;

  int ow0 = m0 % OW; int t0 = m0 / OW; int oh0 = t0 % OH; int n0 = t0 / OH;
  int ow1 = m1 % OW; int t1 = m1 / OW; int oh1 = t1 % OH; int n1 = t1 / OH;
  int ih0a = oh0*S - P, iw0a = ow0*S - P;
  int ih1a = oh1*S - P, iw1a = ow1*S - P;
  const __bf16* base0 = act + (size_t)n0 * IH * IW * Cin;
  const __bf16* base1 = act + (size_t)n1 * IH * IW * Cin;

  int bn = tid >> 2, bc = tid & 3;
  const __bf16* bsrc = Wt + (size_t)(blockIdx.y*64 + bn) * K + bc*8;
  uint32_t blds[2] = {
    (uint32_t)(uintptr_t)&bs[0][bn*BSTRIDE + bc*8],
    (uint32_t)(uintptr_t)&bs[1][bn*BSTRIDE + bc*8]
  };
  int foff = (lane & 15)*BSTRIDE + (lane >> 4)*16;

  v16bf az = zfrag();
  auto loadA = [&](const __bf16* base, int ihb, int iwb, int kb) -> v16bf {
    int k  = kb + koff;
    int kh = k / RC; int rem = k - kh*RC;
    int kw = rem / Cin; int c0 = rem - kw*Cin;
    int ih = ihb + kh, iw = iwb + kw;
    if ((unsigned)ih < (unsigned)IH && (unsigned)iw < (unsigned)IW)
      return ldfrag(base + ((size_t)(ih*IW + iw))*Cin + c0);
    return az;
  };

  v8f acc[8] = { zacc(), zacc(), zacc(), zacc(), zacc(), zacc(), zacc(), zacc() };

  int nt = K / 32;
  async_b128(blds[0], bsrc);
  v16bf a0 = loadA(base0, ih0a, iw0a, 0);
  v16bf a1 = loadA(base1, ih1a, iw1a, 0);
  wait_async0();
  __syncthreads();

  for (int i = 0; i < nt - 1; ++i) {
    async_b128(blds[(i+1)&1], bsrc + (size_t)(i+1)*32);

    const __bf16* bb = &bs[i & 1][foff];
    v16bf b0 = ldsfrag(bb);
    v16bf b1 = ldsfrag(bb + 16*BSTRIDE);
    v16bf b2 = ldsfrag(bb + 32*BSTRIDE);
    v16bf b3 = ldsfrag(bb + 48*BSTRIDE);

    v16bf a0n = loadA(base0, ih0a, iw0a, (i+1)*32);
    v16bf a1n = loadA(base1, ih1a, iw1a, (i+1)*32);

    WMMA_BF16(acc[0], a0, b0); WMMA_BF16(acc[1], a0, b1);
    WMMA_BF16(acc[2], a0, b2); WMMA_BF16(acc[3], a0, b3);
    WMMA_BF16(acc[4], a1, b0); WMMA_BF16(acc[5], a1, b1);
    WMMA_BF16(acc[6], a1, b2); WMMA_BF16(acc[7], a1, b3);
    a0 = a0n; a1 = a1n;

    wait_async0();
    __syncthreads();
  }

  {
    const __bf16* bb = &bs[(nt - 1) & 1][foff];
    v16bf b0 = ldsfrag(bb);
    v16bf b1 = ldsfrag(bb + 16*BSTRIDE);
    v16bf b2 = ldsfrag(bb + 32*BSTRIDE);
    v16bf b3 = ldsfrag(bb + 48*BSTRIDE);
    WMMA_BF16(acc[0], a0, b0); WMMA_BF16(acc[1], a0, b1);
    WMMA_BF16(acc[2], a0, b2); WMMA_BF16(acc[3], a0, b3);
    WMMA_BF16(acc[4], a1, b0); WMMA_BF16(acc[5], a1, b1);
    WMMA_BF16(acc[6], a1, b2); WMMA_BF16(acc[7], a1, b3);
  }

  int rbase0 = mt*32 + (lane >> 4) * 8;
  int col0   = blockIdx.y * 64 + (lane & 15);
  #pragma unroll
  for (int s = 0; s < 4; ++s) {
    int col = col0 + s*16;
    #pragma unroll
    for (int r = 0; r < 8; ++r) {
      out[(size_t)(rbase0 + r)      * Cout + col] = acc[s][r];
      out[(size_t)(rbase0 + 16 + r) * Cout + col] = acc[4 + s][r];
    }
  }
}

// ---------------------------------------------------------------------------
// InstanceNorm + LeakyReLU(0.2): raw NHWC f32 -> act NHWC bf16.
// Conv bias is omitted upstream (cancels exactly in the mean subtraction).
__global__ void k_instnorm(const float* __restrict__ raw, __bf16* __restrict__ act,
                           const float* __restrict__ g, const float* __restrict__ be,
                           int HW, int C) {
  int n = blockIdx.x;
  int c = blockIdx.y * 64 + threadIdx.x;
  const float* p = raw + (size_t)n * HW * C + c;
  float s = 0.f, ss = 0.f;
  for (int i = 0; i < HW; ++i) { float v = p[(size_t)i * C]; s += v; ss += v*v; }
  float mu  = s / (float)HW;
  float var = ss / (float)HW - mu*mu;
  float inv = rsqrtf(var + 1e-5f) * g[c];
  float bb  = be[c];
  __bf16* q = act + (size_t)n * HW * C + c;
  for (int i = 0; i < HW; ++i) {
    float v = (p[(size_t)i*C] - mu) * inv + bb;
    v = v >= 0.f ? v : 0.2f*v;
    q[(size_t)i*C] = (__bf16)v;
  }
}

// Layer-4 variant: emits feat in NCHW-flat order (f32 -> d_out, bf16 -> mb matrix).
__global__ void k_instnorm4(const float* __restrict__ raw, float* __restrict__ feat,
                            __bf16* __restrict__ mbbf,
                            const float* __restrict__ g, const float* __restrict__ be) {
  int n = blockIdx.x;
  int c = blockIdx.y * 64 + threadIdx.x;            // 0..255
  const float* p = raw + (size_t)n * 16 * 256 + c;
  float s = 0.f, ss = 0.f;
  for (int i = 0; i < 16; ++i) { float v = p[i*256]; s += v; ss += v*v; }
  float mu  = s * (1.f/16.f);
  float var = ss * (1.f/16.f) - mu*mu;
  float inv = rsqrtf(var + 1e-5f) * g[c];
  float bb  = be[c];
  for (int i = 0; i < 16; ++i) {
    float v = (p[i*256] - mu) * inv + bb;
    v = v >= 0.f ? v : 0.2f*v;
    feat[(size_t)n*4096 + c*16 + i] = v;
    mbbf[(size_t)n*4160 + c*16 + i] = (__bf16)v;
  }
}

// ---------------------------------------------------------------------------
// Minibatch discrimination: Mm (256 x 3200 f32, [b][o*50+k]).
// o_b[j,o] = sum_i exp(-sum_k |M[i,o,k]-M[j,o,k]|) - 1  -> bf16 into mb[:,4096+o].
__global__ void k_mbd(const float* __restrict__ Mm, __bf16* __restrict__ mbbf) {
  int j = blockIdx.x, tid = threadIdx.x;
  __shared__ float mj[3200];
  __shared__ float red[256];
  for (int t = tid; t < 3200; t += 256) mj[t] = Mm[(size_t)j*3200 + t];
  __syncthreads();
  const float* mi = Mm + (size_t)tid * 3200;
  for (int o = 0; o < 64; ++o) {
    float s = 0.f;
    const float* a = mi + o*50;
    const float* b = mj + o*50;
    #pragma unroll 5
    for (int k = 0; k < 50; ++k) s += fabsf(a[k] - b[k]);
    red[tid] = __expf(-s);
    __syncthreads();
    for (int st = 128; st > 0; st >>= 1) {
      if (tid < st) red[tid] += red[tid + st];
      __syncthreads();
    }
    if (tid == 0) mbbf[(size_t)j*4160 + 4096 + o] = (__bf16)(red[0] - 1.0f);
    __syncthreads();
  }
}

// ---------------------------------------------------------------------------
extern "C" void kernel_launch(void* const* d_in, const int* in_sizes, int n_in,
                              void* d_out, int out_size, void* d_ws, size_t ws_size,
                              hipStream_t stream) {
  (void)in_sizes; (void)n_in; (void)out_size; (void)ws_size;
  const float* x   = (const float*)d_in[0];
  const float* w1  = (const float*)d_in[1];
  const float* g1  = (const float*)d_in[3];
  const float* be1 = (const float*)d_in[4];
  const float* w2  = (const float*)d_in[5];
  const float* g2  = (const float*)d_in[7];
  const float* be2 = (const float*)d_in[8];
  const float* w3  = (const float*)d_in[9];
  const float* g3  = (const float*)d_in[11];
  const float* be3 = (const float*)d_in[12];
  const float* w4  = (const float*)d_in[13];
  const float* g4  = (const float*)d_in[15];
  const float* be4 = (const float*)d_in[16];
  const float* T   = (const float*)d_in[17];
  const float* fcw = (const float*)d_in[18];
  const float* fcb = (const float*)d_in[19];

  char* ws = (char*)d_ws;
  // workspace layout (~160.6 MB, all offsets 256B-aligned)
  constexpr size_t OFF_P   = 0;                         // ping: im2col1 / act1 / act2 / act3
  constexpr size_t OFF_Q   = 50331648;                  // pong: raw1 / raw2 / raw3 / raw4
  constexpr size_t OFF_W1T = 117440512;
  constexpr size_t OFF_W2T = OFF_W1T + 12288;
  constexpr size_t OFF_W3T = OFF_W2T + 409600;
  constexpr size_t OFF_W4T = OFF_W3T + 1638400;
  constexpr size_t OFF_TT  = OFF_W4T + 1179648;
  constexpr size_t OFF_FCW = OFF_TT  + 26214400;
  constexpr size_t OFF_MBB = OFF_FCW + 8320000;
  constexpr size_t OFF_MM  = OFF_MBB + 2129920;

  __bf16* A1    = (__bf16*)(ws + OFF_P);
  __bf16* actP  = (__bf16*)(ws + OFF_P);
  float*  rawQ  = (float*)(ws + OFF_Q);
  __bf16* w1t   = (__bf16*)(ws + OFF_W1T);
  __bf16* w2t   = (__bf16*)(ws + OFF_W2T);
  __bf16* w3t   = (__bf16*)(ws + OFF_W3T);
  __bf16* w4t   = (__bf16*)(ws + OFF_W4T);
  __bf16* Tt    = (__bf16*)(ws + OFF_TT);
  __bf16* fcwbf = (__bf16*)(ws + OFF_FCW);
  __bf16* mbbf  = (__bf16*)(ws + OFF_MBB);
  float*  Mm    = (float*)(ws + OFF_MM);

  float* feat = (float*)d_out;                 // 256*4096
  float* y    = (float*)d_out + 1048576;       // 256*1000

  // ---- weight / tensor conversions ----
  k_wt<<<24,    256, 0, stream>>>(w1, w1t,   3, 5,   96, (size_t)64*96);
  k_wt<<<800,   256, 0, stream>>>(w2, w2t,  64, 5, 1600, (size_t)128*1600);
  k_wt<<<3200,  256, 0, stream>>>(w3, w3t, 128, 5, 3200, (size_t)256*3200);
  k_wt<<<2304,  256, 0, stream>>>(w4, w4t, 256, 3, 2304, (size_t)256*2304);
  k_transT<<<51200, 256, 0, stream>>>(T, Tt);
  k_cvt<<<16250, 256, 0, stream>>>(fcw, fcwbf, (size_t)1000*4160);

  // ---- layer 1: im2col + WMMA GEMM (M=262144, N=64, K=96) ----
  k_im2col1<<<98304, 256, 0, stream>>>(x, A1);
  k_gemm<<<dim3(1024, 1), 256, 0, stream>>>(A1, w1t, rawQ, nullptr, 64, 96, 96, 96);
  k_instnorm<<<dim3(256, 1), 64, 0, stream>>>(rawQ, actP, g1, be1, 1024, 64);

  // ---- layer 2: implicit conv (M=65536, N=128, K=1600) ----
  k_conv<<<dim3(256, 2), 256, 0, stream>>>(actP, w2t, rawQ, 32, 32, 64, 16, 16, 128, 5, 2, 2);
  k_instnorm<<<dim3(256, 2), 64, 0, stream>>>(rawQ, actP, g2, be2, 256, 128);

  // ---- layer 3: implicit conv (M=16384, N=256, K=3200) ----
  k_conv<<<dim3(64, 4), 256, 0, stream>>>(actP, w3t, rawQ, 16, 16, 128, 8, 8, 256, 5, 2, 2);
  k_instnorm<<<dim3(256, 4), 64, 0, stream>>>(rawQ, actP, g3, be3, 64, 256);

  // ---- layer 4: implicit conv (M=4096, N=256, K=2304) ----
  k_conv<<<dim3(16, 4), 256, 0, stream>>>(actP, w4t, rawQ, 8, 8, 256, 4, 4, 256, 3, 2, 1);
  k_instnorm4<<<dim3(256, 4), 64, 0, stream>>>(rawQ, feat, mbbf, g4, be4);

  // ---- M = feat x T  (M=256, N=3200, K=4096, lda=4160, ldb=4096) ----
  k_gemm<<<dim3(1, 50), 256, 0, stream>>>(mbbf, Tt, Mm, nullptr, 3200, 4096, 4160, 4096);

  // ---- minibatch discrimination -> mb[:,4096:4160] ----
  k_mbd<<<256, 256, 0, stream>>>(Mm, mbbf);

  // ---- y = mb x fc_w^T + fc_b  (M=256, N=1000, K=4160, lda=ldb=4160) ----
  k_gemm<<<dim3(1, 16), 256, 0, stream>>>(mbbf, fcwbf, y, fcb, 1000, 4160, 4160, 4160);
}
